// GraphNetworkSimulator_14130442403992
// MI455X (gfx1250) — matrix operand order repaired and checked by reading, not compiled, not run on hardware
//
#include <hip/hip_runtime.h>
#include <hip/hip_bf16.h>
#include <stdint.h>

// ---------------------------------------------------------------------------
// GraphNetworkSimulator on gfx1250 (MI455X), WMMA f16-in/f32-acc MLPs,
// async-LDS weight loads, vectorized f32->f16 staging.
// ---------------------------------------------------------------------------

typedef _Float16 half_t;
typedef __attribute__((ext_vector_type(16))) half_t v16h;
typedef __attribute__((ext_vector_type(4)))  half_t half4;
typedef __attribute__((ext_vector_type(8)))  float  v8f;

#define NNODES 50000
#define NEDGES 800000
#define LAT 32

// ----------------------- CDNA5 async LDS copy helpers -----------------------
// GLOBAL_LOAD_ASYNC_TO_LDS_B128: VDST = per-lane LDS byte address VGPR,
// VADDR = 64-bit global address. Tracked with ASYNCcnt (ISA 10.? / 15.18.3).
__device__ __forceinline__ void async_b128(uint32_t lds_byte, uint64_t gaddr) {
  asm volatile("global_load_async_to_lds_b128 %0, %1, off"
               :: "v"(lds_byte), "v"(gaddr)
               : "memory");
}
__device__ __forceinline__ void wait_async0() {
#if __has_builtin(__builtin_amdgcn_s_wait_asynccnt)
  __builtin_amdgcn_s_wait_asynccnt(0);
#else
  asm volatile("s_wait_asynccnt 0x0" ::: "memory");
#endif
}
// LDS byte address = low 32 bits of the flat pointer (ISA aperture rule).
__device__ __forceinline__ uint32_t lds_addr_of(const void* p) {
  return (uint32_t)(size_t)p;
}
__device__ __forceinline__ void lds_fill_async(void* dst, const void* src,
                                               int bytes, int tid, int nthr) {
  const uint32_t d = lds_addr_of(dst);
  const uint64_t s = (uint64_t)(size_t)src;
  for (int off = tid * 16; off < bytes; off += nthr * 16)
    async_b128(d + (uint32_t)off, s + (uint64_t)off);
}
// Compiler-only fence: hardware keeps same-wave DS ops in order, so the
// wave-local hidden round-trip needs no s_barrier, just no compiler reordering.
__device__ __forceinline__ void lds_compiler_fence() {
  asm volatile("" ::: "memory");
}

// ---------------- WMMA fragment helpers (wave32, 16x16x32 f16) --------------
// A: 16x32 f16 from row-major staging [16][kpad] in LDS.
// ISA 7.12.2: lanes 0-15 -> M=lane, K {0..7,16..23}; lanes 16-31 -> K+8.
__device__ __forceinline__ v16h load_a_frag(const half_t* sx, int lane, int c, int kpad) {
  const int row = lane & 15;
  const int h   = lane >> 4;
  const half_t* p = sx + row * kpad + c * 32;
  v16h a;
#pragma unroll
  for (int e = 0; e < 8; ++e) a[e] = p[8 * h + e];
#pragma unroll
  for (int e = 0; e < 8; ++e) a[8 + e] = p[16 + 8 * h + e];
  return a;
}

// B: 32x16 f16 per N-tile; weights stored transposed [N=32][kpad]; same K order
// as A: lanes 0-15 -> K {0..7,16..23} (col N=lane), 16-31 -> K+8.
__device__ __forceinline__ v16h load_b_frag(const half_t* wT, int lane, int t, int c, int kpad) {
  const int n  = 16 * t + (lane & 15);
  const int kh = lane >> 4;
  const half_t* p = wT + n * kpad + c * 32;
  v16h b;
#pragma unroll
  for (int e = 0; e < 8; ++e) b[e] = p[8 * kh + e];
#pragma unroll
  for (int e = 0; e < 8; ++e) b[8 + e] = p[16 + 8 * kh + e];
  return b;
}

__device__ __forceinline__ v8f wmma_f16(v16h a, v16h b, v8f c) {
  return __builtin_amdgcn_wmma_f32_16x16x32_f16(false, a, false, b, (short)0, c,
                                                false, false);
}

// out(16x32) = relu(X(16xk1pad) @ W1 + b1) @ W2 + b2; per-wave sx/sh staging.
__device__ __forceinline__ void mlp2_tile(const half_t* sx, int k1pad,
                                          const half_t* sW1, const float* sB1,
                                          const half_t* sW2, const float* sB2,
                                          half_t* sh, int lane,
                                          v8f& o0, v8f& o1) {
  v8f a0 = {}, a1 = {};
  const int nch = k1pad >> 5;
  for (int c = 0; c < nch; ++c) {
    v16h a = load_a_frag(sx, lane, c, k1pad);
    a0 = wmma_f16(a, load_b_frag(sW1, lane, 0, c, k1pad), a0);
    a1 = wmma_f16(a, load_b_frag(sW1, lane, 1, c, k1pad), a1);
  }
  const int n  = lane & 15;
  const int mh = (lane >> 4) * 8;
#pragma unroll
  for (int r = 0; r < 8; ++r) {
    float h0 = fmaxf(a0[r] + sB1[n], 0.0f);
    float h1 = fmaxf(a1[r] + sB1[16 + n], 0.0f);
    sh[(mh + r) * 32 + n]      = (half_t)h0;
    sh[(mh + r) * 32 + 16 + n] = (half_t)h1;
  }
  lds_compiler_fence();  // same-wave DS ordering handles the hardware side
  v8f c0 = {}, c1 = {};
  v16h ah = load_a_frag(sh, lane, 0, 32);
  c0 = wmma_f16(ah, load_b_frag(sW2, lane, 0, 0, 32), c0);
  c1 = wmma_f16(ah, load_b_frag(sW2, lane, 1, 0, 32), c1);
#pragma unroll
  for (int r = 0; r < 8; ++r) {
    o0[r] = c0[r] + sB2[n];
    o1[r] = c1[r] + sB2[16 + n];
  }
}

__device__ __forceinline__ half4 cvt4(float4 v) {
  half4 h = {(half_t)v.x, (half_t)v.y, (half_t)v.z, (half_t)v.w};
  return h;
}

// -------------------- weight prep: f32 -> f16 transposed --------------------
struct MlpPtrs { const float *W1, *b1, *W2, *b2; int K1, N2; };
struct PrepArgs { MlpPtrs m[10]; };

// Per MLP j: wT1 at wbase+j*5120 ([32][k1pad] halfs), wT2 at +4096 ([32][32]).
// b1 at bbase+j*64, b2 at +32 (f32, zero-padded).
__global__ __launch_bounds__(256) void prep_weights_kernel(PrepArgs args,
                                                           half_t* wbase,
                                                           float* bbase) {
  const int j = blockIdx.x;
  const MlpPtrs p = args.m[j];
  const int k1pad = (p.K1 <= 32) ? 32 : 128;
  half_t* w1d = wbase + (size_t)j * 5120;
  half_t* w2d = w1d + 4096;
  float*  b1d = bbase + (size_t)j * 64;
  float*  b2d = b1d + 32;
  for (int i = threadIdx.x; i < 32 * k1pad; i += 256) {
    int n = i / k1pad, k = i % k1pad;                // W1 is (K1,32) row-major
    w1d[i] = (k < p.K1) ? (half_t)p.W1[(size_t)k * 32 + n] : (half_t)0.0f;
  }
  for (int i = threadIdx.x; i < 1024; i += 256) {
    int n = i / 32, k = i % 32;                      // W2 is (32,N2) row-major
    w2d[i] = (n < p.N2) ? (half_t)p.W2[(size_t)k * p.N2 + n] : (half_t)0.0f;
  }
  if (threadIdx.x < 32) {
    b1d[threadIdx.x] = p.b1[threadIdx.x];
    b2d[threadIdx.x] = (threadIdx.x < p.N2) ? p.b2[threadIdx.x] : 0.0f;
  }
}

__global__ __launch_bounds__(256) void zero2_kernel(float* a, float* b, int n) {
  for (int i = blockIdx.x * 256 + threadIdx.x; i < n; i += gridDim.x * 256) {
    a[i] = 0.0f;
    b[i] = 0.0f;
  }
}

// --------------------------- encoder kernel --------------------------------
__global__ __launch_bounds__(128) void enc_kernel(const float* __restrict__ in,
                                                  int din, float* __restrict__ out,
                                                  const half_t* w1T, const float* b1,
                                                  const half_t* w2T, const float* b2,
                                                  int rows) {
  __shared__ half_t sW1[32 * 32];
  __shared__ half_t sW2[32 * 32];
  __shared__ float  sB1[32], sB2[32];
  __shared__ half_t sX[4][16 * 32];
  __shared__ half_t sH[4][16 * 32];
  const int tid = threadIdx.x;
  lds_fill_async(sW1, w1T, 2048, tid, 128);
  lds_fill_async(sW2, w2T, 2048, tid, 128);
  if (tid < 32) { sB1[tid] = b1[tid]; sB2[tid] = b2[tid]; }
  const int wave = tid >> 5, lane = tid & 31;
  const int base = blockIdx.x * 64 + wave * 16;
  half_t* sx = sX[wave];
  for (int r = 0; r < 16; ++r) {
    int e = base + r; if (e >= rows) e = rows - 1;
    sx[r * 32 + lane] = (lane < din) ? (half_t)in[(size_t)e * din + lane] : (half_t)0.0f;
  }
  wait_async0();
  __syncthreads();
  v8f o0, o1;
  mlp2_tile(sx, 32, sW1, sB1, sW2, sB2, sH[wave], lane, o0, o1);
  const int n = lane & 15, mh = (lane >> 4) * 8;
#pragma unroll
  for (int r = 0; r < 8; ++r) {
    int e = base + mh + r;
    if (e < rows) {
      out[(size_t)e * LAT + n]      = o0[r];
      out[(size_t)e * LAT + 16 + n] = o1[r];
    }
  }
}

// ---------------------- MP edge update (+ fused scatter) --------------------
__global__ __launch_bounds__(128) void edge_mp_kernel(
    const float* __restrict__ nodes, float* __restrict__ edges,
    float* __restrict__ sent, float* __restrict__ recv,
    const int* __restrict__ senders, const int* __restrict__ receivers,
    const float* __restrict__ g4,
    const half_t* w1T, const float* b1, const half_t* w2T, const float* b2,
    int E) {
  __shared__ half_t sW1[32 * 128];
  __shared__ half_t sW2[32 * 32];
  __shared__ float  sB1[32], sB2[32];
  __shared__ half_t sX[4][16 * 128];
  __shared__ half_t sH[4][16 * 32];
  const int tid = threadIdx.x;
  // warm L2/WGP$ for this block's edge tile (64 rows x 128B)
  __builtin_prefetch(edges + (size_t)blockIdx.x * 64 * LAT + tid * 16, 0, 3);
  lds_fill_async(sW1, w1T, 32 * 128 * 2, tid, 128);
  lds_fill_async(sW2, w2T, 32 * 32 * 2, tid, 128);
  if (tid < 32) { sB1[tid] = b1[tid]; sB2[tid] = b2[tid]; }
  const int wave = tid >> 5, lane = tid & 31;
  const int base = blockIdx.x * 64 + wave * 16;
  half_t* sx = sX[wave];
  // stage concat(edge, nodes[s], nodes[r], g) -> 16x128 f16 tile
  // lane -> segment (lane>>3) and 4-col group ((lane&7)*4): float4 in, half4 out
  const int seg = lane >> 3;
  const int q   = (lane & 7) * 4;
  half4 gtail = {(half_t)0.0f, (half_t)0.0f, (half_t)0.0f, (half_t)0.0f};
  if (seg == 3 && q == 0) {
    half4 g = {(half_t)g4[0], (half_t)g4[1], (half_t)g4[2], (half_t)g4[3]};
    gtail = g;
  }
  for (int r = 0; r < 16; ++r) {
    int e = base + r; if (e >= E) e = E - 1;
    half4 hv = gtail;
    if (seg < 3) {
      const float* src;
      if (seg == 0)      src = edges + (size_t)e * LAT + q;
      else if (seg == 1) src = nodes + (size_t)senders[e] * LAT + q;
      else               src = nodes + (size_t)receivers[e] * LAT + q;
      hv = cvt4(*(const float4*)src);
    }
    *(half4*)(sx + r * 128 + seg * 32 + q) = hv;
  }
  wait_async0();
  __syncthreads();
  v8f o0, o1;
  mlp2_tile(sx, 128, sW1, sB1, sW2, sB2, sH[wave], lane, o0, o1);
  const int n = lane & 15, mh = (lane >> 4) * 8;
#pragma unroll
  for (int r = 0; r < 8; ++r) {
    const int e = base + mh + r;
    if (e < E) {
      const float u0 = o0[r], u1 = o1[r];
      const size_t eo = (size_t)e * LAT;
      edges[eo + n]      += u0;   // residual
      edges[eo + 16 + n] += u1;
      const int s = senders[e], rr = receivers[e];
      atomicAdd(&sent[(size_t)s * LAT + n], u0);        // segment-sum (L2 resident)
      atomicAdd(&sent[(size_t)s * LAT + 16 + n], u1);
      atomicAdd(&recv[(size_t)rr * LAT + n], u0);
      atomicAdd(&recv[(size_t)rr * LAT + 16 + n], u1);
    }
  }
}

// --------------------------- MP node update --------------------------------
__global__ __launch_bounds__(128) void node_mp_kernel(
    float* __restrict__ nodes, const float* __restrict__ sent,
    const float* __restrict__ recv, const float* __restrict__ g4,
    const half_t* w1T, const float* b1, const half_t* w2T, const float* b2,
    int N) {
  __shared__ half_t sW1[32 * 128];
  __shared__ half_t sW2[32 * 32];
  __shared__ float  sB1[32], sB2[32];
  __shared__ half_t sX[4][16 * 128];
  __shared__ half_t sH[4][16 * 32];
  const int tid = threadIdx.x;
  lds_fill_async(sW1, w1T, 32 * 128 * 2, tid, 128);
  lds_fill_async(sW2, w2T, 32 * 32 * 2, tid, 128);
  if (tid < 32) { sB1[tid] = b1[tid]; sB2[tid] = b2[tid]; }
  const int wave = tid >> 5, lane = tid & 31;
  const int base = blockIdx.x * 64 + wave * 16;
  half_t* sx = sX[wave];
  const int seg = lane >> 3;
  const int q   = (lane & 7) * 4;
  half4 gtail = {(half_t)0.0f, (half_t)0.0f, (half_t)0.0f, (half_t)0.0f};
  if (seg == 3 && q == 0) {
    half4 g = {(half_t)g4[0], (half_t)g4[1], (half_t)g4[2], (half_t)g4[3]};
    gtail = g;
  }
  for (int r = 0; r < 16; ++r) {
    int e = base + r; if (e >= N) e = N - 1;
    const size_t eo = (size_t)e * LAT + q;
    half4 hv = gtail;
    if (seg < 3) {
      const float* src = (seg == 0) ? nodes + eo : (seg == 1) ? sent + eo : recv + eo;
      hv = cvt4(*(const float4*)src);
    }
    *(half4*)(sx + r * 128 + seg * 32 + q) = hv;
  }
  wait_async0();
  __syncthreads();
  v8f o0, o1;
  mlp2_tile(sx, 128, sW1, sB1, sW2, sB2, sH[wave], lane, o0, o1);
  const int n = lane & 15, mh = (lane >> 4) * 8;
#pragma unroll
  for (int r = 0; r < 8; ++r) {
    const int e = base + mh + r;
    if (e < N) {
      nodes[(size_t)e * LAT + n]      += o0[r];  // residual
      nodes[(size_t)e * LAT + 16 + n] += o1[r];
    }
  }
}

// ----------------------------- decoder -------------------------------------
__global__ __launch_bounds__(128) void dec_kernel(const float* __restrict__ lat,
                                                  float* __restrict__ out, int dout,
                                                  const half_t* w1T, const float* b1,
                                                  const half_t* w2T, const float* b2,
                                                  int rows) {
  __shared__ half_t sW1[32 * 32];
  __shared__ half_t sW2[32 * 32];
  __shared__ float  sB1[32], sB2[32];
  __shared__ half_t sX[4][16 * 32];
  __shared__ half_t sH[4][16 * 32];
  const int tid = threadIdx.x;
  lds_fill_async(sW1, w1T, 2048, tid, 128);
  lds_fill_async(sW2, w2T, 2048, tid, 128);
  if (tid < 32) { sB1[tid] = b1[tid]; sB2[tid] = b2[tid]; }
  const int wave = tid >> 5, lane = tid & 31;
  const int base = blockIdx.x * 64 + wave * 16;
  half_t* sx = sX[wave];
  const int q = (lane & 7) * 4;
  for (int rr = 0; rr < 4; ++rr) {
    const int row = rr * 4 + (lane >> 3);
    int e = base + row; if (e >= rows) e = rows - 1;
    half4 hv = cvt4(*(const float4*)(lat + (size_t)e * LAT + q));
    *(half4*)(sx + row * 32 + q) = hv;
  }
  wait_async0();
  __syncthreads();
  v8f o0, o1;
  mlp2_tile(sx, 32, sW1, sB1, sW2, sB2, sH[wave], lane, o0, o1);
  const int n = lane & 15, mh = (lane >> 4) * 8;
#pragma unroll
  for (int r = 0; r < 8; ++r) {
    const int e = base + mh + r;
    if (e < rows && n < dout) out[(size_t)e * dout + n] = o0[r];
  }
}

// ============================== launcher ====================================
extern "C" void kernel_launch(void* const* d_in, const int* in_sizes, int n_in,
                              void* d_out, int out_size, void* d_ws, size_t ws_size,
                              hipStream_t stream) {
  (void)in_sizes; (void)n_in; (void)out_size;
  const float* node_features = (const float*)d_in[0];  // 50000x8
  const float* edge_features = (const float*)d_in[1];  // 800000x4
  const float* aux           = (const float*)d_in[2];  // 4
  const int*   senders       = (const int*)d_in[3];    // 800000
  const int*   receivers     = (const int*)d_in[4];    // 800000
  // params pytree leaves (jax tree-flatten, sorted dict keys):
  // dec_edge[4], dec_node[4], enc_edge[4], enc_node[4], mp0.e, mp0.n, ... mp2.n
  const float* P[40];
  for (int i = 0; i < 40; ++i) P[i] = (const float*)d_in[5 + i];

  // ------- ws layout -------
  size_t off = 0;
  auto alloc = [&](size_t bytes) { size_t o = off; off = (off + bytes + 255) & ~(size_t)255; return o; };
  const size_t o_nodes = alloc((size_t)NNODES * LAT * 4);
  const size_t o_edges = alloc((size_t)NEDGES * LAT * 4);
  const size_t o_sent  = alloc((size_t)NNODES * LAT * 4);
  const size_t o_recv  = alloc((size_t)NNODES * LAT * 4);
  const size_t o_wgt   = alloc((size_t)10 * 5120 * 2);
  const size_t o_bias  = alloc((size_t)10 * 64 * 4);
  if (off > ws_size) return;  // workspace too small; nothing sane to do

  char* ws = (char*)d_ws;
  float*  nodes_lat = (float*)(ws + o_nodes);
  float*  edges_lat = (float*)(ws + o_edges);
  float*  sent      = (float*)(ws + o_sent);
  float*  recv      = (float*)(ws + o_recv);
  half_t* wbase     = (half_t*)(ws + o_wgt);
  float*  bbase     = (float*)(ws + o_bias);

  auto wslot = [&](int j) { return wbase + (size_t)j * 5120; };
  auto bslot = [&](int j) { return bbase + (size_t)j * 64; };

  // slots: 0 enc_node, 1 enc_edge, 2..7 mp{0,1,2}.{edge,node}, 8 dec_node, 9 dec_edge
  PrepArgs pa;
  auto set = [&](int slot, int pidx, int K1, int N2) {
    pa.m[slot] = MlpPtrs{P[pidx], P[pidx + 1], P[pidx + 2], P[pidx + 3], K1, N2};
  };
  set(0, 12, 8, 32);    // enc_node
  set(1, 8, 4, 32);     // enc_edge
  for (int s = 0; s < 3; ++s) {
    set(2 + 2 * s, 16 + 8 * s, 100, 32);      // mp[s].edge
    set(3 + 2 * s, 20 + 8 * s, 100, 32);      // mp[s].node
  }
  set(8, 4, 32, 3);     // dec_node
  set(9, 0, 32, 1);     // dec_edge

  prep_weights_kernel<<<10, 256, 0, stream>>>(pa, wbase, bbase);

  const int nblk = (NNODES + 63) / 64;   // 782
  const int eblk = (NEDGES + 63) / 64;   // 12500

  enc_kernel<<<nblk, 128, 0, stream>>>(node_features, 8, nodes_lat,
                                       wslot(0), bslot(0), wslot(0) + 4096, bslot(0) + 32,
                                       NNODES);
  enc_kernel<<<eblk, 128, 0, stream>>>(edge_features, 4, edges_lat,
                                       wslot(1), bslot(1), wslot(1) + 4096, bslot(1) + 32,
                                       NEDGES);
  for (int s = 0; s < 3; ++s) {
    zero2_kernel<<<2048, 256, 0, stream>>>(sent, recv, NNODES * LAT);
    const int je = 2 + 2 * s, jn = 3 + 2 * s;
    edge_mp_kernel<<<eblk, 128, 0, stream>>>(nodes_lat, edges_lat, sent, recv,
                                             senders, receivers, aux,
                                             wslot(je), bslot(je),
                                             wslot(je) + 4096, bslot(je) + 32, NEDGES);
    node_mp_kernel<<<nblk, 128, 0, stream>>>(nodes_lat, sent, recv, aux,
                                             wslot(jn), bslot(jn),
                                             wslot(jn) + 4096, bslot(jn) + 32, NNODES);
  }
  float* out = (float*)d_out;
  dec_kernel<<<nblk, 128, 0, stream>>>(nodes_lat, out, 3,
                                       wslot(8), bslot(8), wslot(8) + 4096, bslot(8) + 32,
                                       NNODES);
  dec_kernel<<<eblk, 128, 0, stream>>>(edges_lat, out + (size_t)NNODES * 3, 1,
                                       wslot(9), bslot(9), wslot(9) + 4096, bslot(9) + 32,
                                       NEDGES);
}
